// Block_88433376625372
// MI455X (gfx1250) — compile-verified, matrix operand order checked by reading, not run on hardware
//
#include <hip/hip_runtime.h>
#include <hip/hip_bf16.h>

// ---------------------------------------------------------------------------
// Transformer block for MI455X (gfx1250, wave32, WMMA).
// All GEMMs use V_WMMA_F32_16X16X32_F16 (f16 in, f32 accumulate).
// B=128 T=256 C=384 H=6 Dh=64 F=1536.
// ---------------------------------------------------------------------------

typedef __attribute__((ext_vector_type(16))) _Float16 v16h;
typedef __attribute__((ext_vector_type(8)))  _Float16 v8h;
typedef __attribute__((ext_vector_type(8)))  float    v8f;

#define DEV __device__ __forceinline__

DEV v8f wmma_f16(v16h a, v16h b, v8f c) {
  // (neg_a, A, neg_b, B, c_mod, C, reuse_a, reuse_b)
  return __builtin_amdgcn_wmma_f32_16x16x32_f16(false, a, false, b, (short)0, c,
                                                false, false);
}

// Load a 16x32 f16 fragment (A-operand layout; B-operand assumed mirrored) from
// a row-major tile with leading dim `ld`. ISA layout (CDNA5 ISA, 16-bit A
// 16x32): lanes 0-15 hold K=0..7 & 16..23; lanes 16-31 hold K=8..15 & 24..31;
// M = lane%16. -> two contiguous 16-byte loads per lane.
DEV v16h load_frag(const _Float16* __restrict__ tile, int ld, int lane) {
  const int m  = lane & 15;
  const int kh = (lane & 16) >> 1;           // 0 or 8
  const _Float16* p = tile + (size_t)m * ld + kh;
  v8h lo = *(const v8h*)p;
  v8h hi = *(const v8h*)(p + 16);
  v16h a;
#pragma unroll
  for (int i = 0; i < 8; ++i) { a[i] = lo[i]; a[i + 8] = hi[i]; }
  return a;
}

// ---------------------------------------------------------------------------
// Weight cast f32 -> f16 with transpose: in (batch,R,C) -> out (batch,C,R).
// ---------------------------------------------------------------------------
__global__ void tcast_kernel(const float* __restrict__ in,
                             _Float16* __restrict__ out, int R, int C) {
  size_t base = (size_t)blockIdx.y * R * C;
  int idx = blockIdx.x * blockDim.x + threadIdx.x;
  if (idx >= R * C) return;
  int r = idx / C, c = idx - r * C;
  out[base + (size_t)c * R + r] = (_Float16)in[base + idx];
}

// ---------------------------------------------------------------------------
// LayerNorm over C=384, one wave per row, f32 math, f16 output.
// ---------------------------------------------------------------------------
__global__ void ln_kernel(const float* __restrict__ x, const float* __restrict__ g,
                          const float* __restrict__ be, _Float16* __restrict__ out) {
  const int C = 384;
  int lane = threadIdx.x & 31;
  size_t row = (size_t)blockIdx.x * 8 + (threadIdx.x >> 5);
  const float* xr = x + row * C;
  float v[12];
  float s = 0.f;
#pragma unroll
  for (int i = 0; i < 12; ++i) { v[i] = xr[lane + 32 * i]; s += v[i]; }
#pragma unroll
  for (int m = 16; m >= 1; m >>= 1) s += __shfl_xor(s, m, 32);
  float mu = s * (1.f / 384.f);
  float var = 0.f;
#pragma unroll
  for (int i = 0; i < 12; ++i) { float d = v[i] - mu; var += d * d; }
#pragma unroll
  for (int m = 16; m >= 1; m >>= 1) var += __shfl_xor(var, m, 32);
  float rs = rsqrtf(var * (1.f / 384.f) + 1e-5f);
  _Float16* orow = out + row * C;
#pragma unroll
  for (int i = 0; i < 12; ++i) {
    int c = lane + 32 * i;
    orow[c] = (_Float16)((v[i] - mu) * rs * g[c] + be[c]);
  }
}

// ---------------------------------------------------------------------------
// Fused QKV: each 32-thread wave computes one 16x16 tile of Q, K and V^T for
// one head, sharing the activation fragment across the three WMMAs.
//   q,k: (B,H,T,Dh) f16     vt: (B,H,Dh,T) f16 (pre-transposed for P*V)
// ---------------------------------------------------------------------------
__global__ void qkv_kernel(const _Float16* __restrict__ h16,
                           const _Float16* __restrict__ wqt,
                           const _Float16* __restrict__ wkt,
                           const _Float16* __restrict__ wvt,
                           _Float16* __restrict__ q, _Float16* __restrict__ k,
                           _Float16* __restrict__ vt) {
  const int Cc = 384, Dh = 64, T = 256, H = 6;
  int tile = blockIdx.x;
  int dt = tile & 3;
  int rest = tile >> 2;
  int hh = rest % H;
  int rt = rest / H;                         // row-tile over B*T
  int lane = threadIdx.x;
  const int n = lane & 15, mh = (lane >> 4) * 8;

  const _Float16* Ab = h16 + (size_t)rt * 16 * Cc;
  const _Float16* Bq = wqt + ((size_t)hh * Dh + dt * 16) * Cc;
  const _Float16* Bk = wkt + ((size_t)hh * Dh + dt * 16) * Cc;
  const _Float16* Bv = wvt + ((size_t)hh * Dh + dt * 16) * Cc;

  v8f cq = {}, ck = {}, cv = {};
  for (int k0 = 0; k0 < Cc; k0 += 32) {
    v16h fh = load_frag(Ab + k0, Cc, lane);
    v16h fq = load_frag(Bq + k0, Cc, lane);
    v16h fk = load_frag(Bk + k0, Cc, lane);
    v16h fv = load_frag(Bv + k0, Cc, lane);
    cq = wmma_f16(fh, fq, cq);               // Q tile  (M=t, N=dh)
    ck = wmma_f16(fh, fk, ck);               // K tile  (M=t, N=dh)
    cv = wmma_f16(fv, fh, cv);               // V^T tile (M=dh, N=t)
  }
  int b = rt >> 4, t0 = (rt & 15) * 16;
  size_t bh = (size_t)b * H + hh;
#pragma unroll
  for (int r = 0; r < 8; ++r) {
    size_t qoff = (bh * T + t0 + mh + r) * Dh + dt * 16 + n;
    q[qoff] = (_Float16)cq[r];
    k[qoff] = (_Float16)ck[r];
    vt[(bh * Dh + dt * 16 + mh + r) * (size_t)T + t0 + n] = (_Float16)cv[r];
  }
}

// ---------------------------------------------------------------------------
// Causal flash attention. One wave per (b,h,16-row query tile). Keys processed
// in chunks of 32; online softmax with per-row stats replicated across the
// 16-lane half-groups; P goes through LDS to become an A-operand fragment.
// ---------------------------------------------------------------------------
__global__ void attn_kernel(const _Float16* __restrict__ q,
                            const _Float16* __restrict__ k,
                            const _Float16* __restrict__ vt,
                            _Float16* __restrict__ attn_out) {
  const int T = 256, Dh = 64, H = 6, C = 384;
  int tile = blockIdx.x;
  int it = tile & 15;
  int bh = tile >> 4;
  int b = bh / H, h = bh - b * H;
  int t0 = it * 16;
  int lane = threadIdx.x;
  const int n = lane & 15, mh = (lane >> 4) * 8;

  const _Float16* qb = q + ((size_t)bh * T + t0) * Dh;
  const _Float16* kb = k + (size_t)bh * T * Dh;
  const _Float16* vb = vt + (size_t)bh * Dh * T;

  v16h aq0 = load_frag(qb, Dh, lane);
  v16h aq1 = load_frag(qb + 32, Dh, lane);

  float mrow[8], lrow[8];
#pragma unroll
  for (int r = 0; r < 8; ++r) { mrow[r] = -__builtin_inff(); lrow[r] = 0.f; }
  v8f O[4];
#pragma unroll
  for (int i = 0; i < 4; ++i) O[i] = (v8f){};

  __shared__ _Float16 Pl[16 * 32];
  const float scale = 0.05103103630798287f;  // 1/sqrt(384) (ref scales by C!)

  for (int s0 = 0; s0 < t0 + 16; s0 += 32) {
    float P0[8], P1[8];
    {
      v8f acc = {};
      v16h b0 = load_frag(kb + (size_t)s0 * Dh, Dh, lane);
      v16h b1 = load_frag(kb + (size_t)s0 * Dh + 32, Dh, lane);
      acc = wmma_f16(aq0, b0, acc);
      acc = wmma_f16(aq1, b1, acc);
#pragma unroll
      for (int r = 0; r < 8; ++r) P0[r] = acc[r] * scale;
    }
    {
      v8f acc = {};
      v16h b0 = load_frag(kb + (size_t)(s0 + 16) * Dh, Dh, lane);
      v16h b1 = load_frag(kb + (size_t)(s0 + 16) * Dh + 32, Dh, lane);
      acc = wmma_f16(aq0, b0, acc);
      acc = wmma_f16(aq1, b1, acc);
#pragma unroll
      for (int r = 0; r < 8; ++r) P1[r] = acc[r] * scale;
    }
#pragma unroll
    for (int r = 0; r < 8; ++r) {
      int trow = t0 + mh + r;
      if (s0 + n > trow)      P0[r] = -__builtin_inff();
      if (s0 + 16 + n > trow) P1[r] = -__builtin_inff();
      float mx = fmaxf(P0[r], P1[r]);
      mx = fmaxf(mx, __shfl_xor(mx, 1, 32));
      mx = fmaxf(mx, __shfl_xor(mx, 2, 32));
      mx = fmaxf(mx, __shfl_xor(mx, 4, 32));
      mx = fmaxf(mx, __shfl_xor(mx, 8, 32));
      float mnew = fmaxf(mrow[r], mx);       // finite: col s0 <= trow always
      float al = expf(mrow[r] - mnew);
      float p0 = expf(P0[r] - mnew);
      float p1 = expf(P1[r] - mnew);
      float rs = p0 + p1;
      rs += __shfl_xor(rs, 1, 32);
      rs += __shfl_xor(rs, 2, 32);
      rs += __shfl_xor(rs, 4, 32);
      rs += __shfl_xor(rs, 8, 32);
      lrow[r] = lrow[r] * al + rs;
      mrow[r] = mnew;
      P0[r] = p0; P1[r] = p1;
      O[0][r] *= al; O[1][r] *= al; O[2][r] *= al; O[3][r] *= al;
    }
    __syncthreads();
#pragma unroll
    for (int r = 0; r < 8; ++r) {            // C/D layout -> row-major in LDS
      Pl[(mh + r) * 32 + n]      = (_Float16)P0[r];
      Pl[(mh + r) * 32 + 16 + n] = (_Float16)P1[r];
    }
    __syncthreads();
    v16h ap = load_frag(Pl, 32, lane);       // A-operand fragment of P (16x32)
#pragma unroll
    for (int nt = 0; nt < 4; ++nt) {
      v16h bv = load_frag(vb + (size_t)(nt * 16) * T + s0, T, lane);
      O[nt] = wmma_f16(ap, bv, O[nt]);
    }
  }
#pragma unroll
  for (int r = 0; r < 8; ++r) {
    float inv = 1.f / lrow[r];
    _Float16* orow = attn_out + ((size_t)b * T + t0 + mh + r) * C + h * Dh;
#pragma unroll
    for (int nt = 0; nt < 4; ++nt) orow[nt * 16 + n] = (_Float16)(O[nt][r] * inv);
  }
}

// ---------------------------------------------------------------------------
// Generic wave-tile GEMM: D(16x16) = A(16xK) * Bt(16xK)^T, plus epilogue.
//  EPI 0: out_f32 = acc + bias + res        (proj + residual)
//  EPI 1: out_f16 = relu(acc + bias)        (FC1)
//  EPI 2: out_f32 += acc + bias             (FC2 + residual accumulate)
// ---------------------------------------------------------------------------
template <int EPI>
__global__ void gemm16(const _Float16* __restrict__ A, const _Float16* __restrict__ Bt,
                       const float* __restrict__ bias, const float* __restrict__ res,
                       float* __restrict__ outf, _Float16* __restrict__ outh,
                       int N, int K) {
  int ntn = N >> 4;
  int tm = blockIdx.x / ntn;
  int tn = blockIdx.x - tm * ntn;
  int lane = threadIdx.x;
  const _Float16* Ab = A + (size_t)tm * 16 * K;
  const _Float16* Bb = Bt + (size_t)tn * 16 * K;
  v8f acc = {};
  for (int k0 = 0; k0 < K; k0 += 32) {
    v16h a = load_frag(Ab + k0, K, lane);
    v16h b = load_frag(Bb + k0, K, lane);
    acc = wmma_f16(a, b, acc);
  }
  const int n = lane & 15, mh = (lane >> 4) * 8;
  const int col = tn * 16 + n;
  const float bv = bias[col];
#pragma unroll
  for (int r = 0; r < 8; ++r) {
    size_t row = (size_t)tm * 16 + mh + r;
    float v = acc[r] + bv;
    if (EPI == 0)      outf[row * N + col] = v + res[row * N + col];
    else if (EPI == 1) outh[row * N + col] = (_Float16)fmaxf(v, 0.f);
    else               outf[row * N + col] += v;
  }
}

// ---------------------------------------------------------------------------
extern "C" void kernel_launch(void* const* d_in, const int* in_sizes, int n_in,
                              void* d_out, int out_size, void* d_ws, size_t ws_size,
                              hipStream_t stream) {
  (void)in_sizes; (void)n_in; (void)out_size; (void)ws_size;
  const float* x   = (const float*)d_in[0];
  const float* Wq  = (const float*)d_in[1];
  const float* Wk  = (const float*)d_in[2];
  const float* Wv  = (const float*)d_in[3];
  const float* Wp  = (const float*)d_in[4];
  const float* bp  = (const float*)d_in[5];
  const float* W1  = (const float*)d_in[6];
  const float* b1  = (const float*)d_in[7];
  const float* W2  = (const float*)d_in[8];
  const float* b2  = (const float*)d_in[9];
  const float* g1  = (const float*)d_in[10];
  const float* be1 = (const float*)d_in[11];
  const float* g2  = (const float*)d_in[12];
  const float* be2 = (const float*)d_in[13];
  float* out = (float*)d_out;

  // Workspace layout (bytes), S = one (B*T, C) f16 buffer = 25,165,824.
  // Total footprint = 6*S + ~3.4MB weights ~= 154.5 MB.
  char* ws = (char*)d_ws;
  const size_t S = (size_t)32768 * 384 * 2;
  _Float16* h16    = (_Float16*)(ws + 0 * S);   // LN1 out
  _Float16* q16    = (_Float16*)(ws + 1 * S);   // (B,H,T,Dh)
  _Float16* k16    = (_Float16*)(ws + 2 * S);   // (B,H,T,Dh)
  _Float16* vt16   = (_Float16*)(ws + 3 * S);   // (B,H,Dh,T)
  _Float16* attn16 = (_Float16*)(ws + 0 * S);   // reuse h16 (dead after QKV)
  _Float16* h2_16  = (_Float16*)(ws + 1 * S);   // reuse q16 (dead after attn)
  _Float16* a1_16  = (_Float16*)(ws + 2 * S);   // 4*S, reuse k16+vt16 span
  char* wbase = ws + 6 * S;
  _Float16* wqt = (_Float16*)(wbase);                        // (H,Dh,C)
  _Float16* wkt = (_Float16*)(wbase + 1 * 294912);
  _Float16* wvt = (_Float16*)(wbase + 2 * 294912);
  _Float16* wpt = (_Float16*)(wbase + 3 * 294912);           // (C,C) = Wp^T
  _Float16* w1t = (_Float16*)(wbase + 4 * 294912);           // (F,C) = W1^T
  _Float16* w2t = (_Float16*)(wbase + 4 * 294912 + 1179648); // (C,F) = W2^T

  // 1) weight casts (f32 -> f16, transposed to Bt layout)
  tcast_kernel<<<dim3(96, 6),   256, 0, stream>>>(Wq, wqt, 384, 64);
  tcast_kernel<<<dim3(96, 6),   256, 0, stream>>>(Wk, wkt, 384, 64);
  tcast_kernel<<<dim3(96, 6),   256, 0, stream>>>(Wv, wvt, 384, 64);
  tcast_kernel<<<dim3(576, 1),  256, 0, stream>>>(Wp, wpt, 384, 384);
  tcast_kernel<<<dim3(2304, 1), 256, 0, stream>>>(W1, w1t, 384, 1536);
  tcast_kernel<<<dim3(2304, 1), 256, 0, stream>>>(W2, w2t, 1536, 384);

  // 2) LN1
  ln_kernel<<<4096, 256, 0, stream>>>(x, g1, be1, h16);
  // 3) QKV projections: (B*T/16)*H*(Dh/16) = 49152 wave-tiles
  qkv_kernel<<<49152, 32, 0, stream>>>(h16, wqt, wkt, wvt, q16, k16, vt16);
  // 4) causal attention: B*H*(T/16) = 12288 wave-tiles
  attn_kernel<<<12288, 32, 0, stream>>>(q16, k16, vt16, attn16);
  // 5) out-proj + bias + residual -> d_out (f32)
  gemm16<0><<<2048 * 24, 32, 0, stream>>>(attn16, wpt, bp, x, out, nullptr, 384, 384);
  // 6) LN2
  ln_kernel<<<4096, 256, 0, stream>>>(out, g2, be2, h2_16);
  // 7) FC1 + ReLU -> f16
  gemm16<1><<<2048 * 96, 32, 0, stream>>>(h2_16, w1t, b1, nullptr, nullptr, a1_16, 1536, 384);
  // 8) FC2 + bias, accumulate into d_out
  gemm16<2><<<2048 * 24, 32, 0, stream>>>(a1_16, w2t, b2, nullptr, out, nullptr, 384, 1536);
}
